// EPMoE_77266461655158
// MI455X (gfx1250) — matrix operand order repaired and checked by reading, not compiled, not run on hardware
//
#include <hip/hip_runtime.h>
#include <hip/hip_bf16.h>

// ---------------- problem constants ----------------
#define T_TOK 8192
#define DH    2048        // hidden
#define DI    2048        // intermediate
#define NE    8           // experts
#define KSEL  2           // experts per token
#define NPAIR (T_TOK*KSEL)      // 16384 token-expert pairs
#define TILE_M 32               // rows per GEMM block
#define MAXTILES (NPAIR/TILE_M + NE)   // 520 worst case

typedef unsigned short u16;
typedef __attribute__((ext_vector_type(16))) __bf16 v16bf;
typedef __attribute__((ext_vector_type(8)))  float  v8f;

union Frag16 { uint4 u[2]; v16bf v; };   // 16 bf16 per lane = 2 x b128

__device__ __forceinline__ u16 f2bf(float f) {
    unsigned u = __float_as_uint(f);
    unsigned r = (u + 0x7FFFu + ((u >> 16) & 1u)) >> 16;   // RNE
    return (u16)r;
}

__device__ __forceinline__ v8f wmma_bf16(const Frag16& a, const Frag16& b, v8f c) {
    return __builtin_amdgcn_wmma_f32_16x16x32_bf16(
        /*neg_a=*/false, a.v, /*neg_b=*/false, b.v,
        /*c_mod=*/(short)0, c, /*reuse_a=*/false, /*reuse_b=*/false);
}

// ---------------- workspace layout ----------------
// [ wsB0 | wsB1 | wsBo ] : bf16 weights pre-swizzled to WMMA B-fragment tiles
//   per matrix: E * (I/16 ntiles) * (D/32 ktiles) * 512 bf16 (32x16 tile)
// [ Meta ]
struct Meta {
    int   counts [NE];
    int   offsets[NE];
    int   cursors[NE];
    int   tileOff[NE + 1];
    int   totalTiles;
    int   _pad[6];
    int   pairE [NPAIR];
    float pairW [NPAIR];
    int   sortedTok[NPAIR];
    float sortedW  [NPAIR];
};
#define PERMAT_US ((size_t)NE * DH * DI)             // ushorts per swizzled matrix
#define META_OFF  (3 * PERMAT_US * sizeof(u16))      // byte offset of Meta in d_ws

// ---------------- kernel: zero output ----------------
__global__ void k_zero_out(float* __restrict__ out) {
    size_t i = (size_t)blockIdx.x * blockDim.x + threadIdx.x;
    ((float4*)out)[i] = make_float4(0.f, 0.f, 0.f, 0.f);
}

// ---------------- kernel: init meta ----------------
__global__ void k_init_meta(Meta* m) {
    int t = threadIdx.x;
    if (t < NE) m->counts[t] = 0;
}

// ---------------- kernel: router (top-2 + softmax) ----------------
__global__ void k_router(const float* __restrict__ logits, Meta* m) {
    int t = blockIdx.x * blockDim.x + threadIdx.x;
    if (t >= T_TOK) return;
    float l[NE];
#pragma unroll
    for (int e = 0; e < NE; ++e) l[e] = logits[t * NE + e];
    int e0 = 0; float b0 = l[0];
#pragma unroll
    for (int e = 1; e < NE; ++e) if (l[e] > b0) { b0 = l[e]; e0 = e; }
    int e1 = -1; float b1 = -3.4e38f;
#pragma unroll
    for (int e = 0; e < NE; ++e) if (e != e0 && l[e] > b1) { b1 = l[e]; e1 = e; }
    float z  = __expf(b1 - b0);            // b1 <= b0
    float w0 = 1.f / (1.f + z);
    float w1 = z   / (1.f + z);
    m->pairE[2 * t]     = e0;  m->pairW[2 * t]     = w0;
    m->pairE[2 * t + 1] = e1;  m->pairW[2 * t + 1] = w1;
    atomicAdd(&m->counts[e0], 1);
    atomicAdd(&m->counts[e1], 1);
}

// ---------------- kernel: tiny scan over E=8 ----------------
__global__ void k_scan(Meta* m) {
    if (threadIdx.x == 0 && blockIdx.x == 0) {
        int off = 0, toff = 0;
        for (int e = 0; e < NE; ++e) {
            m->offsets[e] = off;
            m->cursors[e] = off;
            m->tileOff[e] = toff;
            off  += m->counts[e];
            toff += (m->counts[e] + TILE_M - 1) / TILE_M;
        }
        m->tileOff[NE] = toff;
        m->totalTiles  = toff;
    }
}

// ---------------- kernel: scatter into expert groups ----------------
__global__ void k_scatter(Meta* m) {
    int i = blockIdx.x * blockDim.x + threadIdx.x;
    if (i >= NPAIR) return;
    int e = m->pairE[i];
    int pos = atomicAdd(&m->cursors[e], 1);
    m->sortedTok[pos] = i >> 1;           // token id
    m->sortedW[pos]   = m->pairW[i];
}

// ---------------- kernel: fp32 weights -> bf16 B-fragment swizzle ----------------
// Input matrices viewed as [K=2048 rows, N=2048 cols] row-major.
// Output tile (kt,nt): 32x16, 512 bf16: lane l (n = l&15, kbase = l<16?0:8)
// holds k = kbase+0..7 then kbase+16..23, stored contiguously per lane.
__global__ void k_swizzle(const float* __restrict__ wi0,
                          const float* __restrict__ wi1,
                          const float* __restrict__ wo,
                          u16* __restrict__ wsB) {
    __shared__ float tile[32][129];
    int bid = blockIdx.x;
    int nstrip = bid & 15;  bid >>= 4;    // 16 strips of 128 cols
    int kstrip = bid & 63;  bid >>= 6;    // 64 strips of 32 rows
    int e = bid & 7;        bid >>= 3;
    int mat = bid;                        // 0..2
    const float* W = (mat == 0 ? wi0 : mat == 1 ? wi1 : wo) + (size_t)e * DH * DI;
    u16* dst = wsB + (size_t)mat * PERMAT_US;

    int tid = threadIdx.x;
    // coalesced read of 32x128 fp32 strip
#pragma unroll
    for (int it = 0; it < 16; ++it) {
        int idx = it * 256 + tid;         // 0..4095
        int k = idx >> 7, c = idx & 127;
        tile[k][c] = W[(size_t)(kstrip * 32 + k) * 2048 + nstrip * 128 + c];
    }
    __syncthreads();
    // swizzled write: one wave per 32x16 tile
    int tsub = tid >> 5;                  // 0..7 tile within strip
    int lane = tid & 31;
    int nt   = nstrip * 8 + tsub;         // global 16-col tile
    int nloc = tsub * 16 + (lane & 15);
    int kb   = (lane < 16) ? 0 : 8;
    unsigned pack[8];
#pragma unroll
    for (int j = 0; j < 8; ++j) {
        unsigned lo = f2bf(tile[kb + 2 * j + (j >= 4 ? 8 : 0)][nloc]);
        unsigned hi = f2bf(tile[kb + 2 * j + 1 + (j >= 4 ? 8 : 0)][nloc]);
        pack[j] = lo | (hi << 16);
    }
    size_t base = (((size_t)e * 128 + nt) * 64 + kstrip) * 512 + (size_t)lane * 16;
    uint4* p = (uint4*)(dst + base);
    p[0] = make_uint4(pack[0], pack[1], pack[2], pack[3]);
    p[1] = make_uint4(pack[4], pack[5], pack[6], pack[7]);
}

// ---------------- kernel: fused grouped SwiGLU FFN (bf16 WMMA) ----------------
__global__ __launch_bounds__(256) void k_moe_gemm(const float* __restrict__ x,
                                                  float* __restrict__ out,
                                                  const u16* __restrict__ wsB,
                                                  Meta* __restrict__ m) {
    __shared__ u16 A  [TILE_M][DH + 8];   // 32 x 2056 bf16  (128KB + pad)
    __shared__ u16 INT[TILE_M][DI + 8];   // 32 x 2056 bf16
    __shared__ int   sTok[TILE_M];
    __shared__ float sW  [TILE_M];
    __shared__ int   sInfo[3];

    int tid = threadIdx.x;
    if (tid == 0) {
        int tile = blockIdx.x;
        if (tile >= m->totalTiles) { sInfo[0] = -1; }
        else {
            int e = 0;
            while (e < NE - 1 && tile >= m->tileOff[e + 1]) ++e;
            int rowStart = m->offsets[e] + (tile - m->tileOff[e]) * TILE_M;
            int valid = m->offsets[e] + m->counts[e] - rowStart;
            if (valid > TILE_M) valid = TILE_M;
            sInfo[0] = e; sInfo[1] = rowStart; sInfo[2] = valid;
        }
    }
    __syncthreads();
    if (sInfo[0] < 0) return;
    int e = sInfo[0], rowStart = sInfo[1], valid = sInfo[2];

    if (tid < TILE_M) {
        if (tid < valid) { sTok[tid] = m->sortedTok[rowStart + tid]; sW[tid] = m->sortedW[rowStart + tid]; }
        else             { sTok[tid] = 0;                            sW[tid] = 0.f; }
    }
    __syncthreads();

    // ---- stage gathered activation rows: fp32 global -> bf16 LDS ----
#pragma unroll 4
    for (int it = 0; it < 64; ++it) {
        int idx = it * 256 + tid;         // float4 units: 32*512
        int row = idx >> 9;
        int c4  = idx & 511;
        float4 v = ((const float4*)(x + (size_t)sTok[row] * DH))[c4];
        unsigned p0 = (unsigned)f2bf(v.x) | ((unsigned)f2bf(v.y) << 16);
        unsigned p1 = (unsigned)f2bf(v.z) | ((unsigned)f2bf(v.w) << 16);
        *(uint2*)&A[row][c4 * 4] = make_uint2(p0, p1);
    }
    __syncthreads();

    int lane = tid & 31;
    int wv   = tid >> 5;                  // 8 waves
    int half = (lane < 16) ? 0 : 1;
    int lmod = lane & 15;

    // ================= phase 1: inter = silu(X@wi0) * (X@wi1) =================
    for (int ct = 0; ct < 16; ++ct) {
        int n0 = wv * 256 + ct * 16;
        int nt = n0 >> 4;
        const uint4* B0 = (const uint4*)(wsB + (((size_t)e * 128 + nt) * 64) * 512) + lane * 2;
        const uint4* B1 = (const uint4*)(wsB + PERMAT_US + (((size_t)e * 128 + nt) * 64) * 512) + lane * 2;
        v8f acc00 = {0,0,0,0,0,0,0,0}, acc01 = acc00, acc10 = acc00, acc11 = acc00;
        for (int kt = 0; kt < 64; ++kt) {
            Frag16 b0, b1, a0, a1;
            b0.u[0] = B0[kt * 64]; b0.u[1] = B0[kt * 64 + 1];
            b1.u[0] = B1[kt * 64]; b1.u[1] = B1[kt * 64 + 1];
            int kb = kt * 32 + half * 8;
            a0.u[0] = *(const uint4*)&A[lmod][kb];
            a0.u[1] = *(const uint4*)&A[lmod][kb + 16];
            a1.u[0] = *(const uint4*)&A[16 + lmod][kb];
            a1.u[1] = *(const uint4*)&A[16 + lmod][kb + 16];
            acc00 = wmma_bf16(a0, b0, acc00);
            acc01 = wmma_bf16(a1, b0, acc01);
            acc10 = wmma_bf16(a0, b1, acc10);
            acc11 = wmma_bf16(a1, b1, acc11);
        }
#pragma unroll
        for (int c = 0; c < 8; ++c) {
            int M0 = c + half * 8;
            float h0 = acc00[c], h1 = acc10[c];
            float g  = h0 / (1.f + __expf(-h0));     // silu
            INT[M0][n0 + lmod] = f2bf(g * h1);
            h0 = acc01[c]; h1 = acc11[c];
            g  = h0 / (1.f + __expf(-h0));
            INT[16 + M0][n0 + lmod] = f2bf(g * h1);
        }
    }
    __syncthreads();

    // ================= phase 2: out = inter @ wo, weighted scatter-add =================
    for (int ct = 0; ct < 16; ++ct) {
        int n0 = wv * 256 + ct * 16;
        int nt = n0 >> 4;
        const uint4* Bo = (const uint4*)(wsB + 2 * PERMAT_US + (((size_t)e * 128 + nt) * 64) * 512) + lane * 2;
        v8f acc0 = {0,0,0,0,0,0,0,0}, acc1 = acc0;
        for (int kt = 0; kt < 64; ++kt) {
            Frag16 b, a0, a1;
            b.u[0] = Bo[kt * 64]; b.u[1] = Bo[kt * 64 + 1];
            int kb = kt * 32 + half * 8;
            a0.u[0] = *(const uint4*)&INT[lmod][kb];
            a0.u[1] = *(const uint4*)&INT[lmod][kb + 16];
            a1.u[0] = *(const uint4*)&INT[16 + lmod][kb];
            a1.u[1] = *(const uint4*)&INT[16 + lmod][kb + 16];
            acc0 = wmma_bf16(a0, b, acc0);
            acc1 = wmma_bf16(a1, b, acc1);
        }
#pragma unroll
        for (int c = 0; c < 8; ++c) {
            int M0 = c + half * 8;
            if (M0 < valid)
                atomicAdd(&out[(size_t)sTok[M0] * DH + n0 + lmod], sW[M0] * acc0[c]);
            int M1 = 16 + M0;
            if (M1 < valid)
                atomicAdd(&out[(size_t)sTok[M1] * DH + n0 + lmod], sW[M1] * acc1[c]);
        }
    }
}

// ---------------- host launcher ----------------
extern "C" void kernel_launch(void* const* d_in, const int* in_sizes, int n_in,
                              void* d_out, int out_size, void* d_ws, size_t ws_size,
                              hipStream_t stream) {
    const float* x      = (const float*)d_in[0];
    const float* logits = (const float*)d_in[1];
    const float* wi0    = (const float*)d_in[2];
    const float* wi1    = (const float*)d_in[3];
    const float* wo     = (const float*)d_in[4];
    float* out = (float*)d_out;

    u16*  wsB  = (u16*)d_ws;
    Meta* meta = (Meta*)((char*)d_ws + META_OFF);

    // zero output (16M floats as float4)
    k_zero_out<<<(T_TOK * DH) / (256 * 4), 256, 0, stream>>>(out);
    // reset per-call meta state
    k_init_meta<<<1, 32, 0, stream>>>(meta);
    // pre-swizzle all weights to bf16 fragment layout: 3 mats * 8 experts * 64k * 16n strips
    k_swizzle<<<3 * 8 * 64 * 16, 256, 0, stream>>>(wi0, wi1, wo, wsB);
    // routing
    k_router<<<T_TOK / 256, 256, 0, stream>>>(logits, meta);
    k_scan<<<1, 1, 0, stream>>>(meta);
    k_scatter<<<NPAIR / 256, 256, 0, stream>>>(meta);
    // fused grouped SwiGLU FFN
    k_moe_gemm<<<MAXTILES, 256, 0, stream>>>(x, out, wsB, meta);
}
